// EfficientDualStreamFineGenerator_81217831568036
// MI455X (gfx1250) — compile-verified, matrix operand order checked by reading, not run on hardware
//
#include <hip/hip_runtime.h>
#include <hip/hip_bf16.h>

// ---------------------------------------------------------------------------
// EfficientDualStreamFineGenerator — CDNA5 (gfx1250) implementation, v2.
// All matmuls (projections, conv-as-implicit-GEMM, attention QK^T and PV)
// run on v_wmma_f32_16x16x32_f16 (f16 operands, f32 accumulate).
// v2: compile-time shapes (full unroll), float4 A-fragment loads,
//     LDS-staged B/weight/K/V tiles shared by 8 waves per block.
// ---------------------------------------------------------------------------

typedef _Float16 v16h __attribute__((ext_vector_type(16)));
typedef _Float16 v8h  __attribute__((ext_vector_type(8)));
typedef float    v8f  __attribute__((ext_vector_type(8)));

#define HID   192
#define NPIX  2304          // 48*48
#define BB    2
#define MTOT  4608          // B*N
#define KZ    12
#define ZD    64
#define CD    20
#define POI   21
#define HEADS 4
#define HD    48
#define GRPS  8
#define EDG   36864         // N*16
#define IMGW  48

__device__ __forceinline__ float gelu_f(float x) {
  return 0.5f * x * (1.0f + erff(x * 0.70710678118654752f));
}

// 16-bit A/B fragment K index pattern (CDNA5 ISA 7.12.2):
// lanes 0-15 hold K in {0..7, 16..23}, lanes 16-31 hold K in {8..15, 24..31}.
__device__ __forceinline__ int kpat(int i, int hi) {
  return i + ((i >= 8) ? 8 : 0) + (hi ? 8 : 0);
}

// Pack 16 consecutive-by-lane-layout floats into an f16 fragment register.
__device__ __forceinline__ v16h pack16(float4 f0, float4 f1, float4 f2, float4 f3,
                                       float s) {
  v16h a;
  a[0]=(_Float16)(f0.x*s); a[1]=(_Float16)(f0.y*s); a[2]=(_Float16)(f0.z*s); a[3]=(_Float16)(f0.w*s);
  a[4]=(_Float16)(f1.x*s); a[5]=(_Float16)(f1.y*s); a[6]=(_Float16)(f1.z*s); a[7]=(_Float16)(f1.w*s);
  a[8]=(_Float16)(f2.x*s); a[9]=(_Float16)(f2.y*s); a[10]=(_Float16)(f2.z*s); a[11]=(_Float16)(f2.w*s);
  a[12]=(_Float16)(f3.x*s); a[13]=(_Float16)(f3.y*s); a[14]=(_Float16)(f3.z*s); a[15]=(_Float16)(f3.w*s);
  return a;
}
// Low 8 halves from two float4s, upper 8 halves zero (K-padding).
__device__ __forceinline__ v16h pack8z(float4 f0, float4 f1, float s) {
  v16h a;
  a[0]=(_Float16)(f0.x*s); a[1]=(_Float16)(f0.y*s); a[2]=(_Float16)(f0.z*s); a[3]=(_Float16)(f0.w*s);
  a[4]=(_Float16)(f1.x*s); a[5]=(_Float16)(f1.y*s); a[6]=(_Float16)(f1.z*s); a[7]=(_Float16)(f1.w*s);
#pragma unroll
  for (int i = 8; i < 16; ++i) a[i] = (_Float16)0.f;
  return a;
}

// ---------------------------------------------------------------------------
// WMMA GEMM: C[M,192] = act(A[M,KDIM] @ W[KDIM,192] + bias)
// Split-A: k<SPLIT reads A0 (ld=SPLIT), else A1 (ld=KDIM-SPLIT).
// Block = 256 (8 waves): one 16-col tile of W staged in LDS in fragment
// layout, 8 M-tiles computed against it.  grid = (12, MTOT/128).
// ---------------------------------------------------------------------------
template<int KDIM, int SPLIT, int ACT>
__global__ __launch_bounds__(256)
void k_gemm(const float* __restrict__ A0, const float* __restrict__ A1,
            const float* __restrict__ W, const float* __restrict__ bias,
            float* __restrict__ C) {
  constexpr int NCH = KDIM / 32;
  __shared__ __align__(32) _Float16 Bs[NCH][32][16];
  const int tid = threadIdx.x;
  const int tn  = blockIdx.x;
  // stage W tile (fragment layout), 16-wide coalesced on n
  for (int idx = tid; idx < KDIM * 16; idx += 256) {
    const int n  = idx & 15;
    const int lh = (idx >> 4) & 1;
    const int i  = (idx >> 5) & 15;
    const int ch = idx >> 9;
    const int k  = ch * 32 + kpat(i, lh);
    Bs[ch][lh * 16 + n][i] = (_Float16)W[k * HID + tn * 16 + n];
  }
  __syncthreads();

  const int lane = tid & 31, wv = tid >> 5;
  const int hi = lane >> 4, lm = lane & 15;
  const int mtile = blockIdx.y * 8 + wv;
  const int rowA  = mtile * 16 + lm;

  v8f acc = {0.f,0.f,0.f,0.f,0.f,0.f,0.f,0.f};
#pragma unroll
  for (int ch = 0; ch < NCH; ++ch) {
    const int kb = ch * 32;
    const float* Ap;
    int off;
    if (kb < SPLIT) { Ap = A0; off = rowA * SPLIT + kb; }
    else            { Ap = A1; off = rowA * (KDIM - SPLIT) + (kb - SPLIT); }
    const float4 f0 = *(const float4*)(Ap + off + 8 * hi);
    const float4 f1 = *(const float4*)(Ap + off + 8 * hi + 4);
    const float4 f2 = *(const float4*)(Ap + off + 16 + 8 * hi);
    const float4 f3 = *(const float4*)(Ap + off + 16 + 8 * hi + 4);
    const v16h a = pack16(f0, f1, f2, f3, 1.f);
    const v16h b = *(const v16h*)(&Bs[ch][lane][0]);
    acc = __builtin_amdgcn_wmma_f32_16x16x32_f16(false, a, false, b,
                                                 (short)0, acc, false, false);
  }
#pragma unroll
  for (int r = 0; r < 8; ++r) {
    const int row = mtile * 16 + r + (hi ? 8 : 0);
    const int col = tn * 16 + lm;
    float v = acc[r] + bias[col];
    if (ACT == 1) v = gelu_f(v);
    C[(size_t)row * HID + col] = v;
  }
}

// ---------------------------------------------------------------------------
// 3x3 SAME conv as implicit GEMM over pixel-major activations [B,N,192].
// Weights Wc: [O=192][I=192][3][3]. Block = 256 (8 waves); all 9 taps of one
// 16-cout tile staged in LDS fragment layout (54 KB).  grid = (12, 36).
// ---------------------------------------------------------------------------
__global__ __launch_bounds__(256)
void k_conv3(const float* __restrict__ X, const float* __restrict__ Wc,
             const float* __restrict__ bias, float* __restrict__ Y) {
  __shared__ __align__(32) _Float16 Ws[9][6][32][16];
  const int tid = threadIdx.x;
  const int tn  = blockIdx.x;
  for (int idx = tid; idx < 9 * 6 * 512; idx += 256) {
    const int tap = idx % 9;
    const int c2  = idx / 9;
    const int n   = c2 & 15;
    const int lh  = (c2 >> 4) & 1;
    const int i   = (c2 >> 5) & 15;
    const int ch  = c2 >> 9;
    const int k   = ch * 32 + kpat(i, lh);
    Ws[tap][ch][lh * 16 + n][i] =
        (_Float16)Wc[(size_t)(tn * 16 + n) * (HID * 9) + k * 9 + tap];
  }
  __syncthreads();

  const int lane = tid & 31, wv = tid >> 5;
  const int hi = lane >> 4, lm = lane & 15;
  const int b    = blockIdx.y / 18;
  const int p0   = (blockIdx.y % 18) * 128 + wv * 16;  // 16 | 48: stays in a row
  const int y    = p0 / IMGW;
  const int px   = p0 % IMGW + lm;

  v8f acc = {0.f,0.f,0.f,0.f,0.f,0.f,0.f,0.f};
#pragma unroll
  for (int dy = 0; dy < 3; ++dy) {
#pragma unroll
    for (int dx = 0; dx < 3; ++dx) {
      const int tap = dy * 3 + dx;
      const int yy = y + dy - 1;
      const int xx = px + dx - 1;
      const bool inb = (yy >= 0) & (yy < IMGW) & (xx >= 0) & (xx < IMGW);
      const float msk = inb ? 1.f : 0.f;
      const int yc = min(max(yy, 0), IMGW - 1);
      const int xc = min(max(xx, 0), IMGW - 1);
      const float* arow = X + (size_t)(b * NPIX + yc * IMGW + xc) * HID;
#pragma unroll
      for (int ch = 0; ch < 6; ++ch) {
        const int kb = ch * 32;
        const float4 f0 = *(const float4*)(arow + kb + 8 * hi);
        const float4 f1 = *(const float4*)(arow + kb + 8 * hi + 4);
        const float4 f2 = *(const float4*)(arow + kb + 16 + 8 * hi);
        const float4 f3 = *(const float4*)(arow + kb + 16 + 8 * hi + 4);
        const v16h a = pack16(f0, f1, f2, f3, msk);
        const v16h w = *(const v16h*)(&Ws[tap][ch][lane][0]);
        acc = __builtin_amdgcn_wmma_f32_16x16x32_f16(false, a, false, w,
                                                     (short)0, acc, false, false);
      }
    }
  }
#pragma unroll
  for (int r = 0; r < 8; ++r) {
    const int pm = p0 + r + (hi ? 8 : 0);
    Y[(size_t)(b * NPIX + pm) * HID + tn * 16 + lm] = acc[r] + bias[tn * 16 + lm];
  }
}

// ---------------------------------------------------------------------------
// Flash attention. Block = 256 (8 waves, 8 consecutive 16-row Q tiles);
// each ctx tile's K/V fragments staged once in LDS and consumed by all waves.
// Online softmax with 16-lane shfl_xor reductions; P transposed via LDS.
// grid = (18, HEADS, B).
// ---------------------------------------------------------------------------
__global__ __launch_bounds__(256)
void k_flash(const float* __restrict__ Q, const float* __restrict__ Kc,
             const float* __restrict__ V, float* __restrict__ O) {
  __shared__ __align__(32) _Float16 Kf[2][32][16];
  __shared__ __align__(32) _Float16 Vf[3][32][16];
  __shared__ __align__(32) _Float16 pl[8][16][16];

  const int tid  = threadIdx.x;
  const int lane = tid & 31, wv = tid >> 5;
  const int hi = lane >> 4, lm = lane & 15;
  const int hh = blockIdx.y;
  const int b  = blockIdx.z;
  const int hcol = hh * HD;
  const int it = blockIdx.x * 8 + wv;
  const float rs = 0.14433756729740645f;  // 1/sqrt(48)

  // Q fragments (A role), pre-scaled, HD=48 padded to 64.
  const float* qrow = Q + (size_t)(b * NPIX + it * 16 + lm) * HID + hcol;
  const v16h q0 = pack16(*(const float4*)(qrow + 8 * hi),
                         *(const float4*)(qrow + 8 * hi + 4),
                         *(const float4*)(qrow + 16 + 8 * hi),
                         *(const float4*)(qrow + 16 + 8 * hi + 4), rs);
  const v16h q1 = pack8z(*(const float4*)(qrow + 32 + 8 * hi),
                         *(const float4*)(qrow + 32 + 8 * hi + 4), rs);

  v8f o0 = {0.f,0.f,0.f,0.f,0.f,0.f,0.f,0.f};
  v8f o1 = o0, o2 = o0;
  float mrow[8], lrow[8];
#pragma unroll
  for (int r = 0; r < 8; ++r) { mrow[r] = -1e30f; lrow[r] = 0.f; }

  for (int j = 0; j < NPIX / 16; ++j) {
    // ---- stage K (B role for QK^T) and V (B role for PV) tiles ----
    if (tid < 64) {                     // Kf: (chunk, lane) pairs
      const int c = tid >> 5, l = tid & 31;
      const int lhi = l >> 4, llm = l & 15;
      const float* krow = Kc + (size_t)(b * NPIX + j * 16 + llm) * HID + hcol;
      v16h kf;
      if (c == 0)
        kf = pack16(*(const float4*)(krow + 8 * lhi),
                    *(const float4*)(krow + 8 * lhi + 4),
                    *(const float4*)(krow + 16 + 8 * lhi),
                    *(const float4*)(krow + 16 + 8 * lhi + 4), 1.f);
      else
        kf = pack8z(*(const float4*)(krow + 32 + 8 * lhi),
                    *(const float4*)(krow + 32 + 8 * lhi + 4), 1.f);
      *(v16h*)(&Kf[c][l][0]) = kf;
    } else if (tid < 160) {             // Vf: (tchunk, lane) pairs
      const int t2 = tid - 64;
      const int t = t2 >> 5, l = t2 & 31;
      const int lhi = l >> 4, llm = l & 15;
      v16h vf;
#pragma unroll
      for (int i = 0; i < 8; ++i)
        vf[i] = (_Float16)V[(size_t)(b * NPIX + j * 16 + i + 8 * lhi) * HID +
                            hcol + t * 16 + llm];
#pragma unroll
      for (int i = 8; i < 16; ++i) vf[i] = (_Float16)0.f;
      *(v16h*)(&Vf[t][l][0]) = vf;
    }
    __syncthreads();

    // ---- S = Q K^T ----
    const v16h b0 = *(const v16h*)(&Kf[0][lane][0]);
    const v16h b1 = *(const v16h*)(&Kf[1][lane][0]);
    v8f s = {0.f,0.f,0.f,0.f,0.f,0.f,0.f,0.f};
    s = __builtin_amdgcn_wmma_f32_16x16x32_f16(false, q0, false, b0, (short)0, s, false, false);
    s = __builtin_amdgcn_wmma_f32_16x16x32_f16(false, q1, false, b1, (short)0, s, false, false);

    // ---- online softmax ----
#pragma unroll
    for (int r = 0; r < 8; ++r) {
      const float sm = s[r];
      float mx = sm;
      mx = fmaxf(mx, __shfl_xor(mx, 1));
      mx = fmaxf(mx, __shfl_xor(mx, 2));
      mx = fmaxf(mx, __shfl_xor(mx, 4));
      mx = fmaxf(mx, __shfl_xor(mx, 8));
      const float nm = fmaxf(mrow[r], mx);
      const float sc = expf(mrow[r] - nm);
      const float pv = expf(sm - nm);
      float rsum = pv;
      rsum += __shfl_xor(rsum, 1);
      rsum += __shfl_xor(rsum, 2);
      rsum += __shfl_xor(rsum, 4);
      rsum += __shfl_xor(rsum, 8);
      lrow[r] = lrow[r] * sc + rsum;
      mrow[r] = nm;
      o0[r] *= sc; o1[r] *= sc; o2[r] *= sc;
      pl[wv][r + (hi ? 8 : 0)][lm] = (_Float16)pv;  // C layout -> LDS[M][N]
    }
    // P back in A layout: lane holds row lm, K = 8*hi + (0..7); rest padded.
    // Same-wave LDS store->load is kept in order by hardware; the block-wide
    // barrier below also fences the Kf/Vf reuse across iterations.
    const v8h pr = *(const v8h*)(&pl[wv][lm][8 * hi]);
    v16h ap;
#pragma unroll
    for (int i = 0; i < 8; ++i) ap[i] = pr[i];
#pragma unroll
    for (int i = 8; i < 16; ++i) ap[i] = (_Float16)0.f;

    // ---- out += P @ V ----
    {
      const v16h bv0 = *(const v16h*)(&Vf[0][lane][0]);
      const v16h bv1 = *(const v16h*)(&Vf[1][lane][0]);
      const v16h bv2 = *(const v16h*)(&Vf[2][lane][0]);
      o0 = __builtin_amdgcn_wmma_f32_16x16x32_f16(false, ap, false, bv0, (short)0, o0, false, false);
      o1 = __builtin_amdgcn_wmma_f32_16x16x32_f16(false, ap, false, bv1, (short)0, o1, false, false);
      o2 = __builtin_amdgcn_wmma_f32_16x16x32_f16(false, ap, false, bv2, (short)0, o2, false, false);
    }
    __syncthreads();
  }

#pragma unroll
  for (int r = 0; r < 8; ++r) {
    const float inv = 1.f / lrow[r];
    const int row = it * 16 + r + (hi ? 8 : 0);
    const size_t ob = (size_t)(b * NPIX + row) * HID + hcol;
    O[ob + 0 * 16 + lm] = o0[r] * inv;
    O[ob + 1 * 16 + lm] = o1[r] * inv;
    O[ob + 2 * 16 + lm] = o2[r] * inv;
  }
}

// --------------------------- small scalar kernels --------------------------

__global__ void k_zone_softmax(const float* __restrict__ zl, float* __restrict__ zpf) {
  const int idx = blockIdx.x * blockDim.x + threadIdx.x;
  if (idx >= MTOT) return;
  const int b = idx / NPIX, n = idx % NPIX;
  const size_t base = (size_t)b * KZ * NPIX + n;
  float mx = -1e30f, v[KZ];
  for (int k = 0; k < KZ; ++k) { v[k] = zl[base + (size_t)k * NPIX]; mx = fmaxf(mx, v[k]); }
  float s = 0.f;
  for (int k = 0; k < KZ; ++k) { v[k] = expf(v[k] - mx); s += v[k]; }
  const float inv = 1.f / s;
  for (int k = 0; k < KZ; ++k) zpf[(size_t)idx * KZ + k] = v[k] * inv;
}

__global__ void k_zmean(const float* __restrict__ zpf, float* __restrict__ Z) {
  const int idx = blockIdx.x * blockDim.x + threadIdx.x;
  if (idx >= NPIX * KZ) return;
  const int n = idx / KZ, k = idx % KZ;
  Z[idx] = 0.5f * (zpf[(size_t)n * KZ + k] + zpf[(size_t)(NPIX + n) * KZ + k]);
}

__global__ void k_edge_gate(const int* __restrict__ row, const int* __restrict__ col,
                            const float* __restrict__ aval, const float* __restrict__ Z,
                            float* __restrict__ valb, float* __restrict__ deg) {
  const int e = blockIdx.x * blockDim.x + threadIdx.x;
  if (e >= EDG) return;
  const int r = row[e], c = col[e];
  float d = 0.f;
  for (int k = 0; k < KZ; ++k) d += Z[r * KZ + k] * Z[c * KZ + k];
  const float v = aval[e] * fmaxf(d, 0.f);
  valb[e] = v;
  atomicAdd(&deg[r], v);
}

__global__ void k_edge_norm(const int* __restrict__ row, const float* __restrict__ deg,
                            float* __restrict__ valb) {
  const int e = blockIdx.x * blockDim.x + threadIdx.x;
  if (e >= EDG) return;
  valb[e] /= (deg[row[e]] + 1e-8f);
}

__global__ void k_coarse(const float* __restrict__ cm, const float* __restrict__ W,
                         const float* __restrict__ bias,
                         float* __restrict__ h0, float* __restrict__ hf) {
  const int idx = blockIdx.x * blockDim.x + threadIdx.x;
  if (idx >= MTOT * HID) return;
  const int m = idx / HID, c = idx % HID;
  const int b = m / NPIX, n = m % NPIX;
  const float f0 = cm[(size_t)(b * 2 + 0) * NPIX + n];
  const float f1 = cm[(size_t)(b * 2 + 1) * NPIX + n];
  const float v = f0 * W[c] + f1 * W[HID + c] + bias[c];
  h0[idx] = v;
  hf[idx] = v;
}

__global__ __launch_bounds__(256)
void k_gn_stats(const float* __restrict__ X, float* __restrict__ st) {
  const int bg = blockIdx.x;              // b*8+g
  const int b = bg / GRPS, g = bg % GRPS;
  const int CPG = HID / GRPS;             // 24
  __shared__ float s1[256], s2[256];
  float a = 0.f, q = 0.f;
  for (int i = threadIdx.x; i < CPG * NPIX; i += 256) {
    const int c = g * CPG + i / NPIX, n = i % NPIX;
    const float v = X[(size_t)(b * NPIX + n) * HID + c];
    a += v; q += v * v;
  }
  s1[threadIdx.x] = a; s2[threadIdx.x] = q;
  __syncthreads();
  for (int off = 128; off > 0; off >>= 1) {
    if (threadIdx.x < off) { s1[threadIdx.x] += s1[threadIdx.x + off];
                             s2[threadIdx.x] += s2[threadIdx.x + off]; }
    __syncthreads();
  }
  if (threadIdx.x == 0) {
    const float inv = 1.f / (float)(CPG * NPIX);
    const float mean = s1[0] * inv;
    const float var = s2[0] * inv - mean * mean;
    st[bg * 2 + 0] = mean;
    st[bg * 2 + 1] = rsqrtf(var + 1e-5f);
  }
}

__global__ void k_gn_apply(const float* __restrict__ X, const float* __restrict__ st,
                           const float* __restrict__ gam, const float* __restrict__ bet,
                           float* __restrict__ Y) {
  const int idx = blockIdx.x * blockDim.x + threadIdx.x;
  if (idx >= MTOT * HID) return;
  const int m = idx / HID, c = idx % HID;
  const int b = m / NPIX, g = c / (HID / GRPS);
  const float mean = st[(b * GRPS + g) * 2 + 0];
  const float inv  = st[(b * GRPS + g) * 2 + 1];
  Y[idx] = gelu_f((X[idx] - mean) * inv * gam[c] + bet[c]);
}

__global__ __launch_bounds__(192)
void k_spmm(const int* __restrict__ row, const int* __restrict__ col,
            const float* __restrict__ valb, const float* __restrict__ X,
            float* __restrict__ Y) {
  const int e = blockIdx.x, c = threadIdx.x;
  const int r = row[e], cl = col[e];
  const float v = valb[e];
  for (int b = 0; b < BB; ++b)
    atomicAdd(&Y[(size_t)(b * NPIX + r) * HID + c],
              v * X[(size_t)(b * NPIX + cl) * HID + c]);
}

// mode 0: OUT = LN(X)*g+b ; mode 1: OUT = RES + gelu(LN(X)*g+b)  (in-place ok)
__global__ __launch_bounds__(64)
void k_ln(const float* __restrict__ X, const float* __restrict__ g,
          const float* __restrict__ be, float* __restrict__ OUT,
          const float* __restrict__ RES, int mode) {
  const int row = blockIdx.x, t = threadIdx.x;
  __shared__ float s1[64], s2[64];
  float a = 0.f, q = 0.f;
  for (int c = t; c < HID; c += 64) {
    const float v = X[(size_t)row * HID + c];
    a += v; q += v * v;
  }
  s1[t] = a; s2[t] = q;
  __syncthreads();
  for (int off = 32; off > 0; off >>= 1) {
    if (t < off) { s1[t] += s1[t + off]; s2[t] += s2[t + off]; }
    __syncthreads();
  }
  const float mean = s1[0] / (float)HID;
  const float var  = s2[0] / (float)HID - mean * mean;
  const float inv  = rsqrtf(var + 1e-5f);
  for (int c = t; c < HID; c += 64) {
    float v = (X[(size_t)row * HID + c] - mean) * inv * g[c] + be[c];
    if (mode == 1) v = RES[(size_t)row * HID + c] + gelu_f(v);
    OUT[(size_t)row * HID + c] = v;
  }
}

__global__ void k_blend(float* __restrict__ X, const float* __restrict__ Yn,
                        const float* __restrict__ gate) {
  const int idx = blockIdx.x * blockDim.x + threadIdx.x;
  if (idx >= MTOT * HID) return;
  const float a = gate[0], b = gate[1];
  const float mx = fmaxf(a, b);
  const float e0 = expf(a - mx), e1 = expf(b - mx);
  const float inv = 1.f / (e0 + e1);
  X[idx] = (e0 * inv) * X[idx] + (e1 * inv) * Yn[idx];
}

__global__ __launch_bounds__(128)
void k_mix2(const float* __restrict__ g1, const float* __restrict__ w2,
            const float* __restrict__ b2, const float* __restrict__ hs,
            const float* __restrict__ hf, float* __restrict__ hm) {
  const int m = blockIdx.x * blockDim.x + threadIdx.x;
  if (m >= MTOT) return;
  float d0 = b2[0], d1 = b2[1];
  for (int c = 0; c < HID; ++c) {
    const float v = g1[(size_t)m * HID + c];
    d0 += v * w2[c * 2 + 0];
    d1 += v * w2[c * 2 + 1];
  }
  const float mx = fmaxf(d0, d1);
  const float e0 = expf(d0 - mx), e1 = expf(d1 - mx);
  const float inv = 1.f / (e0 + e1);
  const float ga = e0 * inv, gb = e1 * inv;
  for (int c = 0; c < HID; ++c)
    hm[(size_t)m * HID + c] = ga * hs[(size_t)m * HID + c] + gb * hf[(size_t)m * HID + c];
}

__global__ void k_fusedZ(const float* __restrict__ zt, const float* __restrict__ zw,
                         float* __restrict__ FZ) {
  const int idx = blockIdx.x * blockDim.x + threadIdx.x;
  if (idx >= KZ * HID) return;
  const int k = idx / HID, c = idx % HID;
  float s = 0.f;
  for (int j = 0; j < ZD; ++j) s += zt[k * ZD + j] * zw[j * HID + c];
  FZ[idx] = s;
}

__global__ void k_cond(const float* __restrict__ cv, const float* __restrict__ cw,
                       const float* __restrict__ cb, float* __restrict__ CO) {
  const int idx = blockIdx.x * blockDim.x + threadIdx.x;
  if (idx >= BB * HID) return;
  const int b = idx / HID, c = idx % HID;
  float s = cb[c];
  for (int j = 0; j < CD; ++j) s += cv[b * CD + j] * cw[j * HID + c];
  CO[idx] = s;
}

__global__ void k_zone_cond_add(float* __restrict__ hm, const float* __restrict__ zpf,
                                const float* __restrict__ FZ, const float* __restrict__ zb,
                                const float* __restrict__ CO) {
  const int idx = blockIdx.x * blockDim.x + threadIdx.x;
  if (idx >= MTOT * HID) return;
  const int m = idx / HID, c = idx % HID;
  const int b = m / NPIX;
  float s = 0.f;
  for (int k = 0; k < KZ; ++k) s += zpf[(size_t)m * KZ + k] * FZ[k * HID + c];
  hm[idx] += s + zb[c] + CO[b * HID + c];
}

__global__ __launch_bounds__(128)
void k_head(const float* __restrict__ hm, const float* __restrict__ og,
            const float* __restrict__ ob, const float* __restrict__ hw,
            const float* __restrict__ hb, float* __restrict__ out) {
  const int m = blockIdx.x * blockDim.x + threadIdx.x;
  if (m >= MTOT) return;
  const int b = m / NPIX, n = m % NPIX;
  float a = 0.f, q = 0.f;
  for (int c = 0; c < HID; ++c) {
    const float v = hm[(size_t)m * HID + c];
    a += v; q += v * v;
  }
  const float mean = a / (float)HID;
  const float inv  = rsqrtf(q / (float)HID - mean * mean + 1e-5f);
  float lg[POI];
  for (int p = 0; p < POI; ++p) lg[p] = hb[p];
  for (int c = 0; c < HID; ++c) {
    const float xn = (hm[(size_t)m * HID + c] - mean) * inv * og[c] + ob[c];
    for (int p = 0; p < POI; ++p) lg[p] += xn * hw[c * POI + p];
  }
  float mx = -1e30f;
  for (int p = 0; p < POI; ++p) { lg[p] *= (1.f / 0.7f); mx = fmaxf(mx, lg[p]); }
  float s = 0.f;
  for (int p = 0; p < POI; ++p) { lg[p] = expf(lg[p] - mx); s += lg[p]; }
  const float invs = 1.f / s;
  for (int p = 0; p < POI; ++p)
    out[(size_t)(b * POI + p) * NPIX + n] = lg[p] * invs;
}

// ---------------------------------------------------------------------------
// Host orchestration
// ---------------------------------------------------------------------------
extern "C" void kernel_launch(void* const* d_in, const int* in_sizes, int n_in,
                              void* d_out, int out_size, void* d_ws, size_t ws_size,
                              hipStream_t stream) {
  (void)in_sizes; (void)n_in; (void)out_size; (void)ws_size;
  const float* coarse_map  = (const float*)d_in[0];
  const float* zone_logits = (const float*)d_in[1];
  const float* cond_vec    = (const float*)d_in[2];
  const int*   adj_row     = (const int*)d_in[3];
  const int*   adj_col     = (const int*)d_in[4];
  const float* adj_val     = (const float*)d_in[5];
  // params (make_params insertion order after the 6 arrays)
  const float* coarse_w = (const float*)d_in[6];
  const float* coarse_b = (const float*)d_in[7];
  const float* conv1_w  = (const float*)d_in[8];
  const float* conv1_b  = (const float*)d_in[9];
  const float* gn1_g    = (const float*)d_in[10];
  const float* gn1_b    = (const float*)d_in[11];
  const float* conv2_w  = (const float*)d_in[12];
  const float* conv2_b  = (const float*)d_in[13];
  const float* gn2_g    = (const float*)d_in[14];
  const float* gn2_b    = (const float*)d_in[15];
  const float* bridge_gate[2] = { (const float*)d_in[48], (const float*)d_in[49] };
  const float* mix_w1   = (const float*)d_in[50];
  const float* mix_b1   = (const float*)d_in[51];
  const float* mix_w2   = (const float*)d_in[52];
  const float* mix_b2   = (const float*)d_in[53];
  const float* zone_tab = (const float*)d_in[54];
  const float* zth_w    = (const float*)d_in[55];
  const float* zth_b    = (const float*)d_in[56];
  const float* cond_w   = (const float*)d_in[57];
  const float* cond_b   = (const float*)d_in[58];
  const float* out_g    = (const float*)d_in[59];
  const float* out_b    = (const float*)d_in[60];
  const float* head_w   = (const float*)d_in[61];
  const float* head_b   = (const float*)d_in[62];
  float* out = (float*)d_out;

  const size_t SZ = (size_t)MTOT * HID;
  float* W0 = (float*)d_ws;
  float* h0     = W0 + 0 * SZ;   // h / h_mixed
  float* h_func = W0 + 1 * SZ;
  float* h_spat = W0 + 2 * SZ;
  float* t1     = W0 + 3 * SZ;
  float* t2     = W0 + 4 * SZ;
  float* t3     = W0 + 5 * SZ;
  float* qb     = W0 + 6 * SZ;
  float* kb     = W0 + 7 * SZ;
  float* vb     = W0 + 8 * SZ;
  float* ab     = W0 + 9 * SZ;
  float* g1b    = W0 + 10 * SZ;
  float* sc0    = W0 + 11 * SZ;
  float* small  = W0 + 12 * SZ;
  float* zpf  = small;               small += (size_t)MTOT * KZ;
  float* Zm   = small;               small += (size_t)NPIX * KZ;
  float* valb = small;               small += EDG;
  float* deg  = small;               small += NPIX;
  float* FZ   = small;               small += KZ * HID;
  float* CO   = small;               small += BB * HID;

  const dim3 blk32(32), blk64(64), blk128(128), blk192(192), blk256(256);
  const dim3 gElem((MTOT * HID + 255) / 256);
  const dim3 gGemm(HID / 16, MTOT / 128);       // (12, 36), 8 waves/block
  const dim3 gConv(HID / 16, MTOT / 128);
  const dim3 gAtt(NPIX / 128, HEADS, BB);       // (18, 4, 2), 8 waves/block

  // --- zone probabilities, gated adjacency ---
  k_zone_softmax<<<dim3((MTOT + 255) / 256), blk256, 0, stream>>>(zone_logits, zpf);
  k_zmean<<<dim3((NPIX * KZ + 255) / 256), blk256, 0, stream>>>(zpf, Zm);
  hipMemsetAsync(deg, 0, NPIX * sizeof(float), stream);
  k_edge_gate<<<dim3((EDG + 255) / 256), blk256, 0, stream>>>(adj_row, adj_col, adj_val, Zm, valb, deg);
  k_edge_norm<<<dim3((EDG + 255) / 256), blk256, 0, stream>>>(adj_row, deg, valb);

  // --- coarse projection (K=2: scalar) ---
  k_coarse<<<gElem, blk256, 0, stream>>>(coarse_map, coarse_w, coarse_b, h0, h_func);

  // --- spatial stream: conv -> GN -> GELU, twice ---
  k_conv3<<<gConv, blk256, 0, stream>>>(h0, conv1_w, conv1_b, t1);
  k_gn_stats<<<dim3(BB * GRPS), blk256, 0, stream>>>(t1, CO + BB * HID); // reuse tail? no:
  // (stats buffer) -- use dedicated small slice:
  // NOTE: stats occupy 32 floats right after CO
  k_gn_apply<<<gElem, blk256, 0, stream>>>(t1, CO + BB * HID, gn1_g, gn1_b, t2);
  k_conv3<<<gConv, blk256, 0, stream>>>(t2, conv2_w, conv2_b, t1);
  k_gn_stats<<<dim3(BB * GRPS), blk256, 0, stream>>>(t1, CO + BB * HID);
  k_gn_apply<<<gElem, blk256, 0, stream>>>(t1, CO + BB * HID, gn2_g, gn2_b, h_spat);

  // --- functional stream: 3 residual GCN layers ---
  for (int l = 0; l < 3; ++l) {
    const float* gw  = (const float*)d_in[16 + 4 * l + 0];
    const float* gbi = (const float*)d_in[16 + 4 * l + 1];
    const float* gg  = (const float*)d_in[16 + 4 * l + 2];
    const float* gbe = (const float*)d_in[16 + 4 * l + 3];
    hipMemsetAsync(sc0, 0, SZ * sizeof(float), stream);
    k_spmm<<<dim3(EDG), blk192, 0, stream>>>(adj_row, adj_col, valb, h_func, sc0);
    k_gemm<HID, HID, 0><<<gGemm, blk256, 0, stream>>>(sc0, sc0, gw, gbi, t1);
    k_ln<<<dim3(MTOT), blk64, 0, stream>>>(t1, gg, gbe, h_func, h_func, 1);
  }

  // --- cross-attention bridges ---
  for (int i = 0; i < 2; ++i) {
    const int base = 28 + 10 * i;
    const float* wq = (const float*)d_in[base + 0]; const float* bq = (const float*)d_in[base + 1];
    const float* wk = (const float*)d_in[base + 2]; const float* bk = (const float*)d_in[base + 3];
    const float* wv = (const float*)d_in[base + 4]; const float* bv = (const float*)d_in[base + 5];
    const float* wo = (const float*)d_in[base + 6]; const float* bo = (const float*)d_in[base + 7];
    const float* lg = (const float*)d_in[base + 8]; const float* lb = (const float*)d_in[base + 9];
    for (int dir = 0; dir < 2; ++dir) {
      const float* qin = dir ? h_func : h_spat;
      const float* ctx = dir ? h_spat : h_func;
      float* dst = dir ? t2 : t1;
      k_gemm<HID, HID, 0><<<gGemm, blk256, 0, stream>>>(qin, qin, wq, bq, qb);
      k_gemm<HID, HID, 0><<<gGemm, blk256, 0, stream>>>(ctx, ctx, wk, bk, kb);
      k_gemm<HID, HID, 0><<<gGemm, blk256, 0, stream>>>(ctx, ctx, wv, bv, vb);
      k_flash<<<gAtt, blk256, 0, stream>>>(qb, kb, vb, ab);
      k_gemm<HID, HID, 0><<<gGemm, blk256, 0, stream>>>(ab, ab, wo, bo, t3);
      k_ln<<<dim3(MTOT), blk64, 0, stream>>>(t3, lg, lb, dst, dst, 0);
    }
    k_blend<<<gElem, blk256, 0, stream>>>(h_spat, t1, bridge_gate[i]);
    k_blend<<<gElem, blk256, 0, stream>>>(h_func, t2, bridge_gate[i]);
  }

  // --- mixture gate ---
  k_gemm<2 * HID, HID, 1><<<gGemm, blk256, 0, stream>>>(h_spat, h_func, mix_w1, mix_b1, g1b);
  k_mix2<<<dim3((MTOT + 127) / 128), blk128, 0, stream>>>(g1b, mix_w2, mix_b2, h_spat, h_func, h0);

  // --- zone embedding + conditioning ---
  k_fusedZ<<<dim3((KZ * HID + 255) / 256), blk256, 0, stream>>>(zone_tab, zth_w, FZ);
  k_cond<<<dim3((BB * HID + 255) / 256), blk256, 0, stream>>>(cond_vec, cond_w, cond_b, CO);
  k_zone_cond_add<<<gElem, blk256, 0, stream>>>(h0, zpf, FZ, zth_b, CO);

  // --- output head: LN -> 192x21 -> softmax(T=0.7) -> NCHW transpose ---
  k_head<<<dim3((MTOT + 127) / 128), blk128, 0, stream>>>(h0, out_g, out_b, head_w, head_b, out);
}
